// NodeSubModel_31533649887477
// MI455X (gfx1250) — compile-verified, matrix operand order checked by reading, not run on hardware
//
#include <hip/hip_runtime.h>

typedef __attribute__((ext_vector_type(16))) _Float16 v16h;
typedef __attribute__((ext_vector_type(8)))  float    v8f;

#define WV 8                  // waves per 256-thread block (wave32)
#define MT 2                  // 16-row M-tiles per wave (B-fragment reuse factor)
#define E_EDGES 800000
#define NN 50000
#define NTILES_N (NN / 16)    // 3125 (exact)

#define WMMA(c, a, b) \
  (c) = __builtin_amdgcn_wmma_f32_16x16x32_f16(false, (a), false, (b), (short)0, (c), false, false)

// Load 16 contiguous fp32, convert to one f16 fragment half-row (16 halves).
__device__ __forceinline__ v16h cvt16(const float* __restrict__ p) {
  v16h a;
  const float4* q = (const float4*)p;
#pragma unroll
  for (int i = 0; i < 4; ++i) {
    float4 f = q[i];
    a[4*i+0] = (_Float16)f.x; a[4*i+1] = (_Float16)f.y;
    a[4*i+2] = (_Float16)f.z; a[4*i+3] = (_Float16)f.w;
  }
  return a;
}
__device__ __forceinline__ v16h cvt16s(const float* __restrict__ p, float s) {
  v16h a;
  const float4* q = (const float4*)p;
#pragma unroll
  for (int i = 0; i < 4; ++i) {
    float4 f = q[i];
    a[4*i+0] = (_Float16)(f.x*s); a[4*i+1] = (_Float16)(f.y*s);
    a[4*i+2] = (_Float16)(f.z*s); a[4*i+3] = (_Float16)(f.w*s);
  }
  return a;
}

// fp32 -> f16 weight repack (W1a | W1b | W2a | W2b | Wr), 57344 elements total.
__global__ void cvt_weights(const float* __restrict__ W1a, const float* __restrict__ W1b,
                            const float* __restrict__ W2a, const float* __restrict__ W2b,
                            const float* __restrict__ Wr, _Float16* __restrict__ out) {
  int i = blockIdx.x * blockDim.x + threadIdx.x;
  if      (i < 16384) out[i] = (_Float16)W1a[i];
  else if (i < 24576) out[i] = (_Float16)W1b[i - 16384];
  else if (i < 40960) out[i] = (_Float16)W2a[i - 24576];
  else if (i < 49152) out[i] = (_Float16)W2b[i - 40960];
  else if (i < 57344) out[i] = (_Float16)Wr [i - 49152];
}

// Edge MLP (concat[x[col], e_attr] -> 128 -> relu -> 64) + scatter-add.
// One wave handles MT=2 tiles of 16 edges. E/(16*MT) = 25000 exact -> full EXEC always.
__global__ void __launch_bounds__(256) edge_mlp_scatter(
    const float* __restrict__ x, const int* __restrict__ eidx,
    const float* __restrict__ eattr,
    const _Float16* __restrict__ W1a, const float* __restrict__ b1a,
    const _Float16* __restrict__ W1b, const float* __restrict__ b1b,
    float* __restrict__ sums, float* __restrict__ counts)
{
  __shared__ __align__(32) _Float16 hid[WV][MT * 16 * 128];  // 8KB per wave
  const int wave = threadIdx.x >> 5;
  const int lane = threadIdx.x & 31;
  const int ml = lane & 15;        // row within tile
  const int kh = lane >> 4;        // K-half selector
  const int ntask = E_EDGES / (16 * MT);

  for (int task = blockIdx.x * WV + wave; task < ntask; task += (int)gridDim.x * WV) {
    v16h Af[MT][4];
    int dst[MT];
#pragma unroll
    for (int i = 0; i < MT; ++i) {
      const int e   = task * (16*MT) + i*16 + ml;
      const int col = eidx[E_EDGES + e];
      dst[i]        = eidx[e];
      const float* xr = x     + (size_t)col * 64;
      const float* er = eattr + (size_t)e   * 64;
      Af[i][0] = cvt16(xr + 16*kh);
      Af[i][1] = cvt16(xr + 32 + 16*kh);
      Af[i][2] = cvt16(er + 16*kh);
      Af[i][3] = cvt16(er + 32 + 16*kh);
    }

    // ---- layer 1: rolled over 8 N-tiles, each B fragment feeds MT WMMAs ----
#pragma unroll 1
    for (int t = 0; t < 8; ++t) {
      const float bv = b1a[16*t + ml];
      v8f c0 = {bv,bv,bv,bv,bv,bv,bv,bv};
      v8f c1 = c0;
      const _Float16* wrow = W1a + (size_t)(16*t + ml) * 128 + 16*kh;
#pragma unroll
      for (int kt = 0; kt < 4; ++kt) {
        v16h Bf = *(const v16h*)(wrow + 32*kt);
        WMMA(c0, Af[0][kt], Bf);
        WMMA(c1, Af[1][kt], Bf);
      }
#pragma unroll
      for (int r = 0; r < 8; ++r) {
        const int m = r + 8*kh;
        hid[wave][       m*128 + 16*t + ml] = (_Float16)fmaxf(c0[r], 0.0f);
        hid[wave][2048 + m*128 + 16*t + ml] = (_Float16)fmaxf(c1[r], 0.0f);
      }
    }

    v16h Hf[MT][4];
#pragma unroll
    for (int i = 0; i < MT; ++i)
#pragma unroll
      for (int kt = 0; kt < 4; ++kt)
        Hf[i][kt] = *(const v16h*)&hid[wave][i*2048 + ml*128 + 32*kt + 16*kh];

    // ---- layer 2: rolled over 4 N-tiles, scatter-add via __shfl'd destinations ----
#pragma unroll 1
    for (int t = 0; t < 4; ++t) {
      const float bv = b1b[16*t + ml];
      v8f c0 = {bv,bv,bv,bv,bv,bv,bv,bv};
      v8f c1 = c0;
      const _Float16* wrow = W1b + (size_t)(16*t + ml) * 128 + 16*kh;
#pragma unroll
      for (int kt = 0; kt < 4; ++kt) {
        v16h Bf = *(const v16h*)(wrow + 32*kt);
        WMMA(c0, Hf[0][kt], Bf);
        WMMA(c1, Hf[1][kt], Bf);
      }
#pragma unroll
      for (int r = 0; r < 8; ++r) {
        const int m = r + 8*kh;
        atomicAdd(&sums[(size_t)__shfl(dst[0], m, 32) * 64 + 16*t + ml], c0[r]);
        atomicAdd(&sums[(size_t)__shfl(dst[1], m, 32) * 64 + 16*t + ml], c1[r]);
      }
    }
    if (kh == 0) {
      atomicAdd(&counts[dst[0]], 1.0f);
      atomicAdd(&counts[dst[1]], 1.0f);
    }
  }
}

// Node MLPs: concat[x, agg] -> 128 -> relu -> 64 (=z); then [z, x] @ Wr^T + br.
// MT=2 tiles per wave; last task clamps loads, wave-uniformly skips invalid stores.
__global__ void __launch_bounds__(256) node_mlps(
    const float* __restrict__ x, const float* __restrict__ sums,
    const float* __restrict__ counts,
    const _Float16* __restrict__ W2a, const float* __restrict__ b2a,
    const _Float16* __restrict__ W2b, const float* __restrict__ b2b,
    const _Float16* __restrict__ Wr,  const float* __restrict__ br,
    float* __restrict__ out)
{
  __shared__ __align__(32) _Float16 buf[WV][MT * 16 * 128];
  const int wave = threadIdx.x >> 5;
  const int lane = threadIdx.x & 31;
  const int ml = lane & 15;
  const int kh = lane >> 4;
  const int ntask = (NTILES_N + MT - 1) / MT;   // 1563

  for (int task = blockIdx.x * WV + wave; task < ntask; task += (int)gridDim.x * WV) {
    v16h Af[MT][4];
#pragma unroll
    for (int i = 0; i < MT; ++i) {
      int node = (task*MT + i) * 16 + ml;
      if (node >= NN) node = NN - 1;              // clamp (loads only)
      const float* xr = x    + (size_t)node * 64;
      const float* sr = sums + (size_t)node * 64;
      const float inv = 1.0f / fmaxf(counts[node], 1.0f);
      Af[i][0] = cvt16 (xr + 16*kh);              // K  0..63  : node features
      Af[i][1] = cvt16 (xr + 32 + 16*kh);
      Af[i][2] = cvt16s(sr + 16*kh, inv);         // K 64..127 : agg = sums/count
      Af[i][3] = cvt16s(sr + 32 + 16*kh, inv);
    }

    // ---- node_mlp_2 layer 1 -> relu -> LDS ----
#pragma unroll 1
    for (int t = 0; t < 8; ++t) {
      const float bv = b2a[16*t + ml];
      v8f c0 = {bv,bv,bv,bv,bv,bv,bv,bv};
      v8f c1 = c0;
      const _Float16* wrow = W2a + (size_t)(16*t + ml) * 128 + 16*kh;
#pragma unroll
      for (int kt = 0; kt < 4; ++kt) {
        v16h Bf = *(const v16h*)(wrow + 32*kt);
        WMMA(c0, Af[0][kt], Bf);
        WMMA(c1, Af[1][kt], Bf);
      }
#pragma unroll
      for (int r = 0; r < 8; ++r) {
        const int m = r + 8*kh;
        buf[wave][       m*128 + 16*t + ml] = (_Float16)fmaxf(c0[r], 0.0f);
        buf[wave][2048 + m*128 + 16*t + ml] = (_Float16)fmaxf(c1[r], 0.0f);
      }
    }

    v16h Hf[MT][4];
#pragma unroll
    for (int i = 0; i < MT; ++i)
#pragma unroll
      for (int kt = 0; kt < 4; ++kt)
        Hf[i][kt] = *(const v16h*)&buf[wave][i*2048 + ml*128 + 32*kt + 16*kh];

    // ---- node_mlp_2 layer 2 -> z (16x64 per tile), staged to LDS in A-layout ----
#pragma unroll 1
    for (int t = 0; t < 4; ++t) {
      const float bv = b2b[16*t + ml];
      v8f c0 = {bv,bv,bv,bv,bv,bv,bv,bv};
      v8f c1 = c0;
      const _Float16* wrow = W2b + (size_t)(16*t + ml) * 128 + 16*kh;
#pragma unroll
      for (int kt = 0; kt < 4; ++kt) {
        v16h Bf = *(const v16h*)(wrow + 32*kt);
        WMMA(c0, Hf[0][kt], Bf);
        WMMA(c1, Hf[1][kt], Bf);
      }
#pragma unroll
      for (int r = 0; r < 8; ++r) {
        const int m = r + 8*kh;
        buf[wave][       m*64 + 16*t + ml] = (_Float16)c0[r];
        buf[wave][2048 + m*64 + 16*t + ml] = (_Float16)c1[r];
      }
    }

    v16h A3[MT][4];
#pragma unroll
    for (int i = 0; i < MT; ++i) {
      A3[i][0] = *(const v16h*)&buf[wave][i*2048 + ml*64 + 16*kh];       // K 0..63 : z
      A3[i][1] = *(const v16h*)&buf[wave][i*2048 + ml*64 + 32 + 16*kh];
      A3[i][2] = Af[i][0];                                               // K 64..127 : x
      A3[i][3] = Af[i][1];
    }

    // ---- residual combine: [z,x] @ Wr^T + br -> out ----
#pragma unroll 1
    for (int t = 0; t < 4; ++t) {
      const float bv = br[16*t + ml];
      v8f c0 = {bv,bv,bv,bv,bv,bv,bv,bv};
      v8f c1 = c0;
      const _Float16* wrow = Wr + (size_t)(16*t + ml) * 128 + 16*kh;
#pragma unroll
      for (int kt = 0; kt < 4; ++kt) {
        v16h Bf = *(const v16h*)(wrow + 32*kt);
        WMMA(c0, A3[0][kt], Bf);
        WMMA(c1, A3[1][kt], Bf);
      }
#pragma unroll
      for (int i = 0; i < MT; ++i) {
        const int tile = task*MT + i;
        if (tile < NTILES_N) {                       // wave-uniform guard
          const v8f& c = (i == 0) ? c0 : c1;
#pragma unroll
          for (int r = 0; r < 8; ++r) {
            const int m = r + 8*kh;
            out[(size_t)(tile*16 + m) * 64 + 16*t + ml] = c[r];
          }
        }
      }
    }
  }
}

extern "C" void kernel_launch(void* const* d_in, const int* in_sizes, int n_in,
                              void* d_out, int out_size, void* d_ws, size_t ws_size,
                              hipStream_t stream) {
  const float* x     = (const float*)d_in[0];
  const int*   eidx  = (const int*)  d_in[1];   // int32 (JAX x64 off)
  const float* eattr = (const float*)d_in[2];
  // d_in[3]=u, d_in[4]=batch (unused by reference output path)
  const float* W1a = (const float*)d_in[5];
  const float* b1a = (const float*)d_in[6];
  const float* W1b = (const float*)d_in[7];
  const float* b1b = (const float*)d_in[8];
  const float* W2a = (const float*)d_in[9];
  const float* b2a = (const float*)d_in[10];
  const float* W2b = (const float*)d_in[11];
  const float* b2b = (const float*)d_in[12];
  const float* Wr  = (const float*)d_in[13];
  const float* br  = (const float*)d_in[14];

  // Workspace: [f16 weights: 57344 halves][pad to 128KB][sums N*64 f32][counts N f32]
  _Float16* wh    = (_Float16*)d_ws;
  _Float16* W1a_h = wh;
  _Float16* W1b_h = wh + 16384;
  _Float16* W2a_h = wh + 24576;
  _Float16* W2b_h = wh + 40960;
  _Float16* Wr_h  = wh + 49152;
  float* sums   = (float*)((char*)d_ws + (128 << 10));
  float* counts = sums + (size_t)NN * 64;

  hipMemsetAsync(sums, 0, (size_t)NN * 64 * sizeof(float) + NN * sizeof(float), stream);
  cvt_weights<<<224, 256, 0, stream>>>(W1a, W1b, W2a, W2b, Wr, wh);

  const int etasks = E_EDGES / (16 * MT);                 // 25000
  edge_mlp_scatter<<<(etasks + WV - 1) / WV, 256, 0, stream>>>(
      x, eidx, eattr, W1a_h, b1a, W1b_h, b1b, sums, counts);

  const int ntasks = (NTILES_N + MT - 1) / MT;            // 1563
  node_mlps<<<(ntasks + WV - 1) / WV, 256, 0, stream>>>(
      x, sums, counts, W2a_h, b2a, W2b_h, b2b, Wr_h, br, (float*)d_out);
}
